// GQAttention_3487513444938
// MI455X (gfx1250) — compile-verified, matrix operand order checked by reading, not run on hardware
//
#include <hip/hip_runtime.h>
#include <hip/hip_bf16.h>

// ---------------------------------------------------------------------------
// GQA attention block for MI455X (gfx1250), bf16 WMMA pipeline.
//   B=2, S=2048, D=2048, H=32, H_kv=8, hd=64, G=4, causal, RoPE theta=5e5
// ---------------------------------------------------------------------------

typedef __bf16 bf16;
typedef __attribute__((ext_vector_type(16))) __bf16 v16bf;
typedef __attribute__((ext_vector_type(8)))  __bf16 v8bf;
typedef __attribute__((ext_vector_type(8)))  float  v8f;

#define LN_ROPE_THETA 13.122363377404328f   // ln(500000)

// Load a 16-lane-striped 16x32 bf16 fragment: lane's data is two contiguous
// 8-element runs at p and p+16 (elements), per ISA 16-bit A/B VGPR layout.
__device__ __forceinline__ v16bf ldfrag(const bf16* __restrict__ p) {
  v8bf lo = *(const v8bf*)(p);
  v8bf hi = *(const v8bf*)(p + 16);
  v16bf f;
#pragma unroll
  for (int i = 0; i < 8; ++i) { f[i] = lo[i]; f[i + 8] = hi[i]; }
  return f;
}

__device__ __forceinline__ v8f wmma_bf16(v16bf a, v16bf b, v8f c) {
  return __builtin_amdgcn_wmma_f32_16x16x32_bf16(
      /*neg_a=*/false, a, /*neg_b=*/false, b,
      /*c_mod=*/(short)0, c, /*reuse_a=*/false, /*reuse_b=*/false);
}

// ---------------------------------------------------------------------------
// Kernel 1a: f32 -> bf16 convert (grid-stride)
// ---------------------------------------------------------------------------
__global__ __launch_bounds__(256) void cvt_bf16_kernel(
    const float* __restrict__ in, bf16* __restrict__ out, int n) {
  for (int i = blockIdx.x * blockDim.x + threadIdx.x; i < n;
       i += gridDim.x * blockDim.x)
    out[i] = (bf16)in[i];
}

// ---------------------------------------------------------------------------
// Kernel 1b: transpose + convert weight (Kd x N f32, row-major) -> (N x Kd bf16)
// so GEMM B-fragments become contiguous 16B loads along K.
// ---------------------------------------------------------------------------
__global__ __launch_bounds__(256) void transpose_cvt_kernel(
    const float* __restrict__ in, bf16* __restrict__ out, int Kd, int N) {
  int idx = blockIdx.x * blockDim.x + threadIdx.x;
  if (idx >= Kd * N) return;
  int k = idx / N, n = idx % N;
  out[(size_t)n * Kd + k] = (bf16)in[idx];
}

// ---------------------------------------------------------------------------
// Kernel 2/4: WMMA GEMM.  C[M,N] = A[M,K] * BT[N,K]^T, bf16 in / f32 acc.
// One wave computes a 32x64 tile: 2 A frags + 4 B frags -> 8 WMMAs per K-step
// (B panel traffic halved vs a 16-row tile). Optional fused RoPE epilogue
// (pairs adjacent feature columns via lane^1 shuffle in the C-layout).
// ---------------------------------------------------------------------------
template <bool ROPE, bool OUTF32>
__global__ __launch_bounds__(256) void wmma_gemm_kernel(
    const bf16* __restrict__ A, const bf16* __restrict__ BT,
    float* __restrict__ Cf, bf16* __restrict__ Cb,
    int M, int N, int K, int posMask) {
  const int lane = threadIdx.x & 31;
  const int wid  = blockIdx.x * (blockDim.x >> 5) + (threadIdx.x >> 5);
  const int ntn  = N >> 6;                 // 64-wide N strips
  if (wid >= (M >> 5) * ntn) return;       // 32-row M strips
  const int tm = wid / ntn, tn = wid % ntn;
  const int half = lane >> 4, ln = lane & 15;

  const bf16* arow0 = A + (size_t)(tm * 32 + ln) * K + half * 8;
  const bf16* arow1 = arow0 + (size_t)16 * K;
  const bf16* brow[4];
#pragma unroll
  for (int t = 0; t < 4; ++t)
    brow[t] = BT + (size_t)(tn * 64 + t * 16 + ln) * K + half * 8;

  v8f acc[2][4];
#pragma unroll
  for (int blk = 0; blk < 2; ++blk)
#pragma unroll
    for (int t = 0; t < 4; ++t) acc[blk][t] = v8f{};

  for (int k0 = 0; k0 < K; k0 += 32) {
    // pull next K-panel toward the caches (global_prefetch_b8)
    __builtin_prefetch(arow0 + k0 + 256, 0, 1);
    __builtin_prefetch(brow[0] + k0 + 256, 0, 1);
    v16bf a0 = ldfrag(arow0 + k0);
    v16bf a1 = ldfrag(arow1 + k0);
#pragma unroll
    for (int t = 0; t < 4; ++t) {
      v16bf bt = ldfrag(brow[t] + k0);
      acc[0][t] = wmma_bf16(a0, bt, acc[0][t]);
      acc[1][t] = wmma_bf16(a1, bt, acc[1][t]);
    }
  }

#pragma unroll
  for (int blk = 0; blk < 2; ++blk) {
#pragma unroll
    for (int t = 0; t < 4; ++t) {
      const int ng = tn * 64 + t * 16 + ln;  // global column (feature)
#pragma unroll
      for (int r = 0; r < 8; ++r) {
        const int mg = tm * 32 + blk * 16 + r + 8 * half;  // global row (token)
        float v = acc[blk][t][r];
        if (ROPE) {
          float partner = __shfl_xor(v, 1, 32);     // (even,odd) feature pair
          const int d = ng & 63;                    // dim within head (hd=64)
          const float pos = (float)(mg & posMask);  // position within sequence
          const float ang =
              pos * __expf(-LN_ROPE_THETA * (float)(d & ~1) * (1.0f / 64.0f));
          float sn, cs;
          __sincosf(ang, &sn, &cs);
          v = (d & 1) ? (v * cs + partner * sn) : (v * cs - partner * sn);
        }
        if (OUTF32) Cf[(size_t)mg * N + ng] = v;
        else        Cb[(size_t)mg * N + ng] = (bf16)v;
      }
    }
  }
}

// ---------------------------------------------------------------------------
// Kernel 3: causal GQA flash attention. One wave per (batch, head, 16-q tile).
//   S = Q Kt^T via 4x v_wmma per 32-key tile, online softmax (lane-group
//   reductions), P transposed C-layout -> A-layout through LDS, then P*V.
// ---------------------------------------------------------------------------
__global__ __launch_bounds__(256) void flash_attn_kernel(
    const bf16* __restrict__ Q, const bf16* __restrict__ Kt,
    const bf16* __restrict__ V, bf16* __restrict__ Z, int Bn) {
  constexpr int S = 2048, H = 32, HD = 64;
  constexpr int QROW = H * HD;   // 2048
  constexpr int KROW = 8 * HD;   // 512 (H_kv * hd)

  __shared__ __align__(16) bf16 lds[8][16 * 32];  // per-wave P-transpose tile

  const int lane  = threadIdx.x & 31;
  const int wslot = threadIdx.x >> 5;
  const int wid   = blockIdx.x * 8 + wslot;
  const int qtiles = S / 16;
  if (wid >= Bn * H * qtiles) return;

  const int qt = wid % qtiles;
  const int h  = (wid / qtiles) % H;
  const int b  = wid / (qtiles * H);
  const int kv = h >> 2;                  // G = 4
  const int half = lane >> 4, ln = lane & 15;

  // Q fragments (16 queries x 64 dims), loaded once.
  const bf16* qbase = Q + (size_t)(b * S + qt * 16 + ln) * QROW + h * HD + half * 8;
  const v16bf aQ0 = ldfrag(qbase);        // d = 0..31
  const v16bf aQ1 = ldfrag(qbase + 32);   // d = 32..63

  v8f acc[4] = {v8f{}, v8f{}, v8f{}, v8f{}};
  float mrow[8], lrow[8];
#pragma unroll
  for (int r = 0; r < 8; ++r) { mrow[r] = -3.0e38f; lrow[r] = 0.0f; }

  const int nkt = ((qt + 1) * 16 + 31) >> 5;  // key tiles touched (causal)
  const float scale = 0.125f;                 // 1/sqrt(64)

  for (int kt = 0; kt < nkt; ++kt) {
    // ---- scores: two 16x16 tiles covering 32 keys --------------------------
    v8f s[2] = {v8f{}, v8f{}};
#pragma unroll
    for (int t = 0; t < 2; ++t) {
      const bf16* kb =
          Kt + (size_t)(b * S + kt * 32 + t * 16 + ln) * KROW + kv * HD + half * 8;
      s[t] = wmma_bf16(aQ0, ldfrag(kb), s[t]);
      s[t] = wmma_bf16(aQ1, ldfrag(kb + 32), s[t]);
    }

    // ---- online softmax (rows = VGPR idx + half-wave; cols = lanes) --------
    float p0v[8], p1v[8], alpha[8];
#pragma unroll
    for (int r = 0; r < 8; ++r) {
      const int row = qt * 16 + r + 8 * half;
      const int c0  = kt * 32 + ln, c1 = c0 + 16;
      float s0 = (c0 > row) ? -3.0e38f : s[0][r] * scale;
      float s1 = (c1 > row) ? -3.0e38f : s[1][r] * scale;
      float mx = fmaxf(s0, s1);
#pragma unroll
      for (int off = 8; off; off >>= 1) mx = fmaxf(mx, __shfl_xor(mx, off, 32));
      const float mnew = fmaxf(mrow[r], mx);
      const float al   = __expf(mrow[r] - mnew);
      const float p0   = __expf(s0 - mnew);
      const float p1   = __expf(s1 - mnew);
      float ps = p0 + p1;
#pragma unroll
      for (int off = 8; off; off >>= 1) ps += __shfl_xor(ps, off, 32);
      lrow[r] = lrow[r] * al + ps;
      mrow[r] = mnew;
      alpha[r] = al;
      p0v[r] = p0; p1v[r] = p1;
    }
#pragma unroll
    for (int sub = 0; sub < 4; ++sub)
#pragma unroll
      for (int r = 0; r < 8; ++r) acc[sub][r] *= alpha[r];

    // ---- P: C-layout -> A-layout via LDS ----------------------------------
    bf16* lb = &lds[wslot][0];
#pragma unroll
    for (int r = 0; r < 8; ++r) {
      const int m = r + 8 * half;
      lb[m * 32 + ln]      = (bf16)p0v[r];
      lb[m * 32 + 16 + ln] = (bf16)p1v[r];
    }
    asm volatile("s_wait_dscnt 0" ::: "memory");
    const v16bf aP = ldfrag(lb + ln * 32 + half * 8);

    // ---- P (16x32) * V (32 keys x 64 dims) --------------------------------
#pragma unroll
    for (int sub = 0; sub < 4; ++sub) {
      const bf16* vb =
          V + (size_t)(b * S + kt * 32) * KROW + kv * HD + sub * 16 + ln;
      v16bf bv;
#pragma unroll
      for (int j = 0; j < 16; ++j) {
        const int kk = half * 8 + j + ((j >= 8) ? 8 : 0);
        bv[j] = vb[(size_t)kk * KROW];
      }
      acc[sub] = wmma_bf16(aP, bv, acc[sub]);
    }
  }

  // ---- z = acc / l --------------------------------------------------------
#pragma unroll
  for (int sub = 0; sub < 4; ++sub) {
    const int ng = h * HD + sub * 16 + ln;
#pragma unroll
    for (int r = 0; r < 8; ++r) {
      const int row = qt * 16 + r + 8 * half;
      Z[(size_t)(b * S + row) * QROW + ng] = (bf16)(acc[sub][r] / lrow[r]);
    }
  }
}

// ---------------------------------------------------------------------------
// Host-side orchestration
// ---------------------------------------------------------------------------
extern "C" void kernel_launch(void* const* d_in, const int* in_sizes, int n_in,
                              void* d_out, int out_size, void* d_ws, size_t ws_size,
                              hipStream_t stream) {
  (void)in_sizes; (void)n_in; (void)out_size; (void)ws_size;
  constexpr int B = 2, S = 2048, D = 2048, H = 32, HKV = 8, HD = 64;
  constexpr int M = B * S;           // 4096 rows
  constexpr int NQ = H * HD;         // 2048
  constexpr int NKV = HKV * HD;      // 512

  const float* x  = (const float*)d_in[0];
  // d_in[1] = position_ids (arange, derived analytically on device)
  const float* Wq = (const float*)d_in[2];
  const float* Wk = (const float*)d_in[3];
  const float* Wv = (const float*)d_in[4];
  const float* Wo = (const float*)d_in[5];
  float* out = (float*)d_out;

  char* ws = (char*)d_ws;
  size_t off = 0;
  auto alloc = [&](size_t bytes) { char* p = ws + off; off += (bytes + 255) & ~(size_t)255; return p; };
  bf16* xbf  = (bf16*)alloc((size_t)M * D * 2);
  bf16* WqT  = (bf16*)alloc((size_t)NQ * D * 2);
  bf16* WkT  = (bf16*)alloc((size_t)NKV * D * 2);
  bf16* WvT  = (bf16*)alloc((size_t)NKV * D * 2);
  bf16* WoT  = (bf16*)alloc((size_t)D * D * 2);
  bf16* Qbf  = (bf16*)alloc((size_t)M * NQ * 2);
  bf16* Kbf  = (bf16*)alloc((size_t)M * NKV * 2);
  bf16* Vbf  = (bf16*)alloc((size_t)M * NKV * 2);
  bf16* Zbf  = (bf16*)alloc((size_t)M * D * 2);

  // 1) converts / weight transposes
  cvt_bf16_kernel<<<4096, 256, 0, stream>>>(x, xbf, M * D);
  transpose_cvt_kernel<<<(D * NQ + 255) / 256, 256, 0, stream>>>(Wq, WqT, D, NQ);
  transpose_cvt_kernel<<<(D * NKV + 255) / 256, 256, 0, stream>>>(Wk, WkT, D, NKV);
  transpose_cvt_kernel<<<(D * NKV + 255) / 256, 256, 0, stream>>>(Wv, WvT, D, NKV);
  transpose_cvt_kernel<<<(D * D + 255) / 256, 256, 0, stream>>>(Wo, WoT, D, D);

  // 2) projections (+ fused RoPE on Q and K); 1 wave = 32x64 tile
  {
    int waves = (M / 32) * (NQ / 64);   // 4096
    wmma_gemm_kernel<true, false><<<waves / 8, 256, 0, stream>>>(
        xbf, WqT, nullptr, Qbf, M, NQ, D, S - 1);
  }
  {
    int waves = (M / 32) * (NKV / 64);  // 1024
    wmma_gemm_kernel<true, false><<<waves / 8, 256, 0, stream>>>(
        xbf, WkT, nullptr, Kbf, M, NKV, D, S - 1);
    wmma_gemm_kernel<false, false><<<waves / 8, 256, 0, stream>>>(
        xbf, WvT, nullptr, Vbf, M, NKV, D, S - 1);
  }

  // 3) causal GQA flash attention
  {
    int waves = B * H * (S / 16);  // 8192
    flash_attn_kernel<<<waves / 8, 256, 0, stream>>>(Qbf, Kbf, Vbf, Zbf, B);
  }

  // 4) output projection -> f32
  {
    int waves = (M / 32) * (D / 64);    // 4096
    wmma_gemm_kernel<false, true><<<waves / 8, 256, 0, stream>>>(
        Zbf, WoT, out, nullptr, M, D, D, S - 1);
  }
}
